// Multi_Head_Attention_With_RoPE_20495583936566
// MI455X (gfx1250) — compile-verified, matrix operand order checked by reading, not run on hardware
//
#include <hip/hip_runtime.h>
#include <hip/hip_bf16.h>
#include <stdint.h>

#define BB 2
#define TT 2048
#define CC 1024
#define HH 16
#define DD 64

typedef __attribute__((ext_vector_type(16))) __bf16 v16bf;
typedef __attribute__((ext_vector_type(8)))  float  v8f;

union Frag {
  v16bf v;
  uint4 q[2];
  unsigned u[8];
};

__device__ inline unsigned short f2b(float x) {
  unsigned u = __float_as_uint(x);
  return (unsigned short)((u + 0x7FFFu + ((u >> 16) & 1u)) >> 16);
}
__device__ inline unsigned pack2(float a, float b) {
  return (unsigned)f2b(a) | ((unsigned)f2b(b) << 16);
}
__device__ inline v8f zero8() {
  v8f z = {0.f, 0.f, 0.f, 0.f, 0.f, 0.f, 0.f, 0.f};
  return z;
}

// Load a 16x32 bf16 "A-style" fragment from a row-major [rows][ld] bf16 matrix.
// A layout: lanes 0-15 -> rows 0..15, K = 0..7 (vgpr0-3) & 16..23 (vgpr4-7);
//           lanes 16-31 -> same rows,  K = 8..15      &      24..31.
// The same loader also produces B-style fragments when applied to the
// transpose-stored matrix (lane = column, contiguous along K).
__device__ inline v16bf load_frag(const unsigned short* __restrict__ base,
                                  int row_base, int k_base, int ld) {
  const int lane = threadIdx.x & 31;
  const unsigned short* p = base + (size_t)(row_base + (lane & 15)) * ld
                                 + k_base + ((lane >> 4) << 3);
  Frag f;
  f.q[0] = *(const uint4*)(p);
  f.q[1] = *(const uint4*)(p + 16);
  return f.v;
}

// gfx1250 async global->LDS copy (16 bytes per lane) + counter waits.
__device__ inline void async_copy16(unsigned lds_addr, const void* g) {
  unsigned long long ga = (unsigned long long)(uintptr_t)g;
  asm volatile("global_load_async_to_lds_b128 %0, %1, off"
               :: "v"(lds_addr), "v"(ga) : "memory");
}
__device__ inline void wait_async0() { asm volatile("s_wait_asynccnt 0x0" ::: "memory"); }
__device__ inline void wait_ds0()    { asm volatile("s_wait_dscnt 0x0" ::: "memory"); }

#define WMMA_BF16(A, B, C) \
  __builtin_amdgcn_wmma_f32_16x16x32_bf16(false, (A), false, (B), (short)0, (C), false, false)

// ---------------------------------------------------------------------------
// Kernel 0: fp32 -> bf16 (RNE)
// ---------------------------------------------------------------------------
__global__ void cvt_bf16_kernel(const float* __restrict__ in,
                                unsigned short* __restrict__ out, int n) {
  int i = blockIdx.x * blockDim.x + threadIdx.x;
  if (i < n) out[i] = f2b(in[i]);
}

// ---------------------------------------------------------------------------
// Kernel 1: QKV projection GEMM + fused RoPE.
// One wave -> 16x64 output tile of q/k/v (blockIdx.z selects the matrix).
// A-tile (x rows) staged in LDS via async copy; weight B-fragments direct.
// Q,K written bf16 [B,H,T,D]; V written transposed bf16 [B,H,D,T].
// ---------------------------------------------------------------------------
__global__ __launch_bounds__(32) void qkv_rope_kernel(
    const unsigned short* __restrict__ xb,
    const unsigned short* __restrict__ wq,
    const unsigned short* __restrict__ wk,
    const unsigned short* __restrict__ wv,
    unsigned short* __restrict__ Qr,
    unsigned short* __restrict__ Kr,
    unsigned short* __restrict__ Vt) {
  __shared__ unsigned short tileA[16 * 32];
  const int lane  = threadIdx.x;
  const int mbase = blockIdx.x * 16;
  const int nbase = blockIdx.y * 64;
  const int sel   = blockIdx.z;
  const unsigned short* __restrict__ W = (sel == 0) ? wq : ((sel == 1) ? wk : wv);

  v8f acc[4];
#pragma unroll
  for (int j = 0; j < 4; ++j) acc[j] = zero8();

  const unsigned ldsbase = (unsigned)(uintptr_t)&tileA[0];

  for (int kc = 0; kc < CC; kc += 32) {
    wait_ds0();  // prior LDS reads done before overwrite
#pragma unroll
    for (int c = 0; c < 2; ++c) {
      unsigned off = (unsigned)(lane + 32 * c) * 16u;  // byte offset in 1KB tile
      unsigned row = off >> 6;                          // 64 B per row
      unsigned col = (off & 63u) >> 1;                  // ushort column
      async_copy16(ldsbase + off,
                   xb + (size_t)(mbase + (int)row) * CC + kc + (int)col);
    }
    wait_async0();

    Frag a;
    const uint4* lp =
        (const uint4*)&tileA[(lane & 15) * 32 + ((lane >> 4) << 3)];
    a.q[0] = lp[0];
    a.q[1] = lp[2];

#pragma unroll
    for (int j = 0; j < 4; ++j) {
      v16bf b = load_frag(W, nbase + j * 16, kc, CC);
      acc[j] = WMMA_BF16(a.v, b, acc[j]);
    }
  }

  // Epilogue: C layout lanes0-15: N=lane, M=r; lanes16-31: N=lane-16, M=r+8.
  const int hi = lane >> 4, ln = lane & 15;
#pragma unroll
  for (int j = 0; j < 4; ++j) {
#pragma unroll
    for (int r = 0; r < 8; ++r) {
      const int m = mbase + r + hi * 8;
      const int n = nbase + j * 16 + ln;
      const int b = m >> 11, t = m & (TT - 1);
      const int h = n >> 6,  d = n & 63;
      float v = acc[j][r];
      if (sel == 2) {
        Vt[((size_t)(b * HH + h) * DD + d) * TT + t] = f2b(v);
      } else {
        // RoPE: pair partner (d^1) lives in adjacent lane of the fragment.
        float prt = __shfl_xor(v, 1);
        float ang = (float)t * __expf((float)(d >> 1) * -0.28782313662425572f);
        float cs = __cosf(ang), sn = __sinf(ang);
        float o = (d & 1) ? (prt * sn + v * cs) : (v * cs - prt * sn);
        unsigned short* dst = (sel == 0) ? Qr : Kr;
        dst[((size_t)(b * HH + h) * TT + t) * DD + d] = f2b(o);
      }
    }
  }
}

// ---------------------------------------------------------------------------
// Kernel 2: causal flash attention, one wave per (b, h, 16 query rows).
// Computes S^T = K·Q^T so lane == query index -> per-lane online softmax,
// then O^T += V^T · P^T with an in-lane C-layout -> B-layout repack of P.
// ---------------------------------------------------------------------------
__global__ __launch_bounds__(32) void flash_kernel(
    const unsigned short* __restrict__ Qr,
    const unsigned short* __restrict__ Kr,
    const unsigned short* __restrict__ Vt,
    unsigned short* __restrict__ Ab) {
  const int lane  = threadIdx.x;
  const int qbase = blockIdx.x * 16;
  const int h     = blockIdx.y;
  const int b     = blockIdx.z;
  const unsigned short* Qh = Qr + (size_t)(b * HH + h) * TT * DD;
  const unsigned short* Kh = Kr + (size_t)(b * HH + h) * TT * DD;
  const unsigned short* Vh = Vt + (size_t)(b * HH + h) * DD * TT;  // [D][T]

  // Q^T B-fragments (lane = query column), two K-chunks over D=64.
  v16bf qb0 = load_frag(Qh, qbase, 0, DD);
  v16bf qb1 = load_frag(Qh, qbase, 32, DD);

  v8f ot[4];
#pragma unroll
  for (int j = 0; j < 4; ++j) ot[j] = zero8();

  const int hi = lane >> 4;
  const int q  = qbase + (lane & 15);
  float mrun = -3.0e38f, lrun = 0.f;

  const int nkt = (qbase + 47) >> 5;  // causal: only k-tiles touching k <= q+15
  for (int kt = 0; kt < nkt; ++kt) {
    const int k0 = kt * 32;
    v8f s[2];
#pragma unroll
    for (int c = 0; c < 2; ++c) {
      v16bf ka0 = load_frag(Kh, k0 + c * 16, 0, DD);
      v16bf ka1 = load_frag(Kh, k0 + c * 16, 32, DD);
      v8f z = zero8();
      z = WMMA_BF16(ka0, qb0, z);
      z = WMMA_BF16(ka1, qb1, z);
      s[c] = z;
    }

    // scale + causal mask + row (=lane) max
    float mx = mrun;
#pragma unroll
    for (int c = 0; c < 2; ++c) {
#pragma unroll
      for (int r = 0; r < 8; ++r) {
        float v = s[c][r] * 0.125f;                 // 1/sqrt(64)
        int kk = k0 + c * 16 + r + hi * 8;
        v = (kk <= q) ? v : -3.0e38f;
        s[c][r] = v;
        mx = fmaxf(mx, v);
      }
    }
    mx = fmaxf(mx, __shfl_xor(mx, 16));  // combine the two k-halves

    float fac = __expf(mrun - mx);
    mrun = mx;

    // exp + repack C-layout (lane=q, vgpr=k) into P^T B-fragment (in-lane).
    float sum = 0.f;
    Frag pb;
#pragma unroll
    for (int c = 0; c < 2; ++c) {
#pragma unroll
      for (int j2 = 0; j2 < 4; ++j2) {
        float p0 = __expf(s[c][2 * j2]     - mx);
        float p1 = __expf(s[c][2 * j2 + 1] - mx);
        sum += p0 + p1;
        pb.u[c * 4 + j2] = pack2(p0, p1);
      }
    }
    sum += __shfl_xor(sum, 16);
    lrun = lrun * fac + sum;

#pragma unroll
    for (int j = 0; j < 4; ++j) {
#pragma unroll
      for (int r = 0; r < 8; ++r) ot[j][r] *= fac;  // per-lane rescale (lane=q)
    }

    // O^T += V^T(16d x 32k) x P^T(32k x 16q), 4 d-tiles.
#pragma unroll
    for (int j = 0; j < 4; ++j) {
      v16bf va = load_frag(Vh, j * 16, k0, TT);
      ot[j] = WMMA_BF16(va, pb.v, ot[j]);
    }
  }

  const float inv = 1.0f / lrun;
#pragma unroll
  for (int j = 0; j < 4; ++j) {
#pragma unroll
    for (int r = 0; r < 8; ++r) {
      int d = j * 16 + r + hi * 8;
      Ab[((size_t)b * TT + q) * CC + h * DD + d] = f2b(ot[j][r] * inv);
    }
  }
}

// ---------------------------------------------------------------------------
// Kernel 3: output projection (attn [4096x1024] bf16 x Wo^T), f32 out.
// ---------------------------------------------------------------------------
__global__ __launch_bounds__(32) void oproj_kernel(
    const unsigned short* __restrict__ ab,
    const unsigned short* __restrict__ wo,
    float* __restrict__ out) {
  __shared__ unsigned short tileA[16 * 32];
  const int lane  = threadIdx.x;
  const int mbase = blockIdx.x * 16;
  const int nbase = blockIdx.y * 64;

  v8f acc[4];
#pragma unroll
  for (int j = 0; j < 4; ++j) acc[j] = zero8();

  const unsigned ldsbase = (unsigned)(uintptr_t)&tileA[0];

  for (int kc = 0; kc < CC; kc += 32) {
    wait_ds0();
#pragma unroll
    for (int c = 0; c < 2; ++c) {
      unsigned off = (unsigned)(lane + 32 * c) * 16u;
      unsigned row = off >> 6;
      unsigned col = (off & 63u) >> 1;
      async_copy16(ldsbase + off,
                   ab + (size_t)(mbase + (int)row) * CC + kc + (int)col);
    }
    wait_async0();

    Frag a;
    const uint4* lp =
        (const uint4*)&tileA[(lane & 15) * 32 + ((lane >> 4) << 3)];
    a.q[0] = lp[0];
    a.q[1] = lp[2];

#pragma unroll
    for (int j = 0; j < 4; ++j) {
      v16bf bfr = load_frag(wo, nbase + j * 16, kc, CC);
      acc[j] = WMMA_BF16(a.v, bfr, acc[j]);
    }
  }

  const int hi = lane >> 4, ln = lane & 15;
#pragma unroll
  for (int j = 0; j < 4; ++j) {
#pragma unroll
    for (int r = 0; r < 8; ++r) {
      const int m = mbase + r + hi * 8;
      const int n = nbase + j * 16 + ln;
      out[(size_t)m * CC + n] = acc[j][r];
    }
  }
}

// ---------------------------------------------------------------------------
// Launch: ws layout (bytes):
//   [0,8M)   xb bf16      [8M,10M) Wq bf16   [10M,12M) Wk   [12M,14M) Wv
//   [14M,16M) Wo bf16     [16M,24M) Q rope   [24M,32M) K rope
//   [32M,40M) V^T bf16    [40M,48M) attn bf16
// ---------------------------------------------------------------------------
extern "C" void kernel_launch(void* const* d_in, const int* in_sizes, int n_in,
                              void* d_out, int out_size, void* d_ws, size_t ws_size,
                              hipStream_t stream) {
  (void)in_sizes; (void)n_in; (void)out_size; (void)ws_size;
  const float* x  = (const float*)d_in[0];
  const float* Wq = (const float*)d_in[1];
  const float* Wk = (const float*)d_in[2];
  const float* Wv = (const float*)d_in[3];
  const float* Wo = (const float*)d_in[4];

  char* ws = (char*)d_ws;
  unsigned short* xb  = (unsigned short*)(ws + (size_t)0);
  unsigned short* wqb = (unsigned short*)(ws + ((size_t)8  << 20));
  unsigned short* wkb = (unsigned short*)(ws + ((size_t)10 << 20));
  unsigned short* wvb = (unsigned short*)(ws + ((size_t)12 << 20));
  unsigned short* wob = (unsigned short*)(ws + ((size_t)14 << 20));
  unsigned short* Qr  = (unsigned short*)(ws + ((size_t)16 << 20));
  unsigned short* Kr  = (unsigned short*)(ws + ((size_t)24 << 20));
  unsigned short* Vt  = (unsigned short*)(ws + ((size_t)32 << 20));
  unsigned short* Ab  = (unsigned short*)(ws + ((size_t)40 << 20));

  const int nx = BB * TT * CC;   // 4194304
  const int nw = CC * CC;        // 1048576
  cvt_bf16_kernel<<<(nx + 255) / 256, 256, 0, stream>>>(x,  xb,  nx);
  cvt_bf16_kernel<<<(nw + 255) / 256, 256, 0, stream>>>(Wq, wqb, nw);
  cvt_bf16_kernel<<<(nw + 255) / 256, 256, 0, stream>>>(Wk, wkb, nw);
  cvt_bf16_kernel<<<(nw + 255) / 256, 256, 0, stream>>>(Wv, wvb, nw);
  cvt_bf16_kernel<<<(nw + 255) / 256, 256, 0, stream>>>(Wo, wob, nw);

  dim3 g1(BB * TT / 16, CC / 64, 3);
  qkv_rope_kernel<<<g1, 32, 0, stream>>>(xb, wqb, wkb, wvb, Qr, Kr, Vt);

  dim3 g2(TT / 16, HH, BB);
  flash_kernel<<<g2, 32, 0, stream>>>(Qr, Kr, Vt, Ab);

  dim3 g3(BB * TT / 16, CC / 64);
  oproj_kernel<<<g3, 32, 0, stream>>>(Ab, wob, (float*)d_out);
}
